// AttentionCell_4398046511572
// MI455X (gfx1250) — matrix operand order repaired and checked by reading, not compiled, not run on hardware
//
#include <hip/hip_runtime.h>
#include <hip/hip_bf16.h>
#include <math.h>

#define TSZ 256
#define BSZ 256
#define NIN 512
#define NH  512
#define NEMB 256
#define NX  (NIN + NEMB)   /* 768  */
#define N3H (3 * NH)       /* 1536 */

typedef __attribute__((ext_vector_type(16))) __bf16 v16bf;
typedef __attribute__((ext_vector_type(8)))  float  v8f;
typedef __attribute__((ext_vector_type(4)))  int    v4i;

// ---------------------------------------------------------------------------
// Optional gfx1250 async global->LDS path (ASYNCcnt-tracked), with fallback.
// Probe result: builtin exists; arg0 is v4i in addrspace(1) ("__device__").
// ---------------------------------------------------------------------------
#if defined(__has_builtin)
#if __has_builtin(__builtin_amdgcn_global_load_async_to_lds_b128)
#define HAVE_ASYNC_LDS 1
#endif
#endif

#ifdef HAVE_ASYNC_LDS
typedef __attribute__((address_space(1))) v4i gbl_v4i;   // global (device) v4i
typedef __attribute__((address_space(3))) v4i lds_v4i;   // LDS (shared) v4i
__device__ inline void async_wait0() {
#if __has_builtin(__builtin_amdgcn_s_wait_asynccnt)
    __builtin_amdgcn_s_wait_asynccnt(0);
#else
    asm volatile("s_wait_asynccnt 0x0" ::: "memory");
#endif
}
#endif

// ---------------------------------------------------------------------------
// WMMA helpers (CDNA5 wave32, 16x16x32 bf16 -> f32)
// A layout (16x32, MxK): lane L holds row m=L%16; elems are two contiguous
//   8-wide K chunks at k0 + (L/16)*8 and k0 + (L/16)*8 + 16.
// B layout (32x16, KxN): lane L holds col n=L%16; elems are two contiguous
//   8-wide K chunks at k0 + (L/16)*8 and k0 + (L/16)*8 + 16.
// C/D layout (16x16): lane L holds col n=L%16, rows m = (L/16)*8 + r.
// ---------------------------------------------------------------------------
__device__ inline v8f wmma_bf16(v16bf a, v16bf b, v8f c) {
    return __builtin_amdgcn_wmma_f32_16x16x32_bf16(
        /*neg_a=*/false, a, /*neg_b=*/false, b,
        /*c_mod=*/(short)0, c, /*reuse_a=*/false, /*reuse_b=*/false);
}

__device__ inline v16bf cvt16(const float* lo, const float* hi) {
    v16bf r;
#pragma unroll
    for (int e = 0; e < 8; ++e) r[e] = (__bf16)lo[e];
#pragma unroll
    for (int e = 0; e < 8; ++e) r[8 + e] = (__bf16)hi[e];
    return r;
}

// A fragment from fp32 (global or LDS), row-major with leading dim ldk.
__device__ inline v16bf load_a_frag_f32(const float* __restrict__ A, int ldk,
                                        int row0, int k0, int lane) {
    const float* p = A + (size_t)(row0 + (lane & 15)) * ldk + k0 + ((lane >> 4) << 3);
    return cvt16(p, p + 16);
}

// B fragment from fp32 global weight W[N,K] row-major (so B = W^T).
__device__ inline v16bf load_b_frag_f32(const float* __restrict__ W, int ldk,
                                        int col0, int k0, int lane) {
    const float* p = W + (size_t)(col0 + (lane & 15)) * ldk + k0 + ((lane >> 4) << 3);
    return cvt16(p, p + 16);
}

// ---------------------------------------------------------------------------
// Generic C[M,N] = A[M,K] @ W[N,K]^T + bias[N]   (fp32 in/out, bf16 WMMA core)
// 256 threads = 8 waves; block tile 32 rows x 512 cols; each wave: 2x4 tiles.
// Inner loop: preload 4 B frags, then 8 back-to-back WMMAs (no interleaved
// VALU -> single WMMA->VALU hazard window per K-step).  grid=(M/32, N/512).
// ---------------------------------------------------------------------------
__global__ void __launch_bounds__(256)
gemm_at_wt(const float* __restrict__ A, const float* __restrict__ W,
           const float* __restrict__ bias, float* __restrict__ C,
           int N, int K) {
    const int lane = threadIdx.x & 31;
    const int wave = threadIdx.x >> 5;
    const int m0 = blockIdx.x * 32;
    const int n0 = blockIdx.y * 512 + wave * 64;

    v8f acc[2][4] = {};
    for (int k0 = 0; k0 < K; k0 += 32) {
        v16bf a0 = load_a_frag_f32(A, K, m0,      k0, lane);
        v16bf a1 = load_a_frag_f32(A, K, m0 + 16, k0, lane);
        v16bf b0 = load_b_frag_f32(W, K, n0,      k0, lane);
        v16bf b1 = load_b_frag_f32(W, K, n0 + 16, k0, lane);
        v16bf b2 = load_b_frag_f32(W, K, n0 + 32, k0, lane);
        v16bf b3 = load_b_frag_f32(W, K, n0 + 48, k0, lane);
        acc[0][0] = wmma_bf16(a0, b0, acc[0][0]);
        acc[1][0] = wmma_bf16(a1, b0, acc[1][0]);
        acc[0][1] = wmma_bf16(a0, b1, acc[0][1]);
        acc[1][1] = wmma_bf16(a1, b1, acc[1][1]);
        acc[0][2] = wmma_bf16(a0, b2, acc[0][2]);
        acc[1][2] = wmma_bf16(a1, b2, acc[1][2]);
        acc[0][3] = wmma_bf16(a0, b3, acc[0][3]);
        acc[1][3] = wmma_bf16(a1, b3, acc[1][3]);
    }

    const int nc = lane & 15, mh = lane >> 4;
#pragma unroll
    for (int mt = 0; mt < 2; ++mt) {
#pragma unroll
        for (int nt = 0; nt < 4; ++nt) {
            const int n = n0 + nt * 16 + nc;
            const float bv = bias[n];
#pragma unroll
            for (int r = 0; r < 8; ++r) {
                const int m = m0 + mt * 16 + mh * 8 + r;
                C[(size_t)m * N + n] = acc[mt][nt][r] + bv;
            }
        }
    }
}

// ---------------------------------------------------------------------------
// Fused attention scores:
//   emit[m] = sum_h tanh( (feats @ w_i2h^T)[m,h] + h'[b,h] ) * w_score[h]
// Block = 32 rows (one t, 32 consecutive b) x all 512 h.  feats tile staged
// fp32 into LDS (async global->LDS when available), converted to bf16 during
// fragment reads.  w_i2h (1MB fp32, L2-resident) read as B frags per wave.
// grid = T*B/32 = 2048, 256 threads, dynamic LDS.
// ---------------------------------------------------------------------------
#define SF_LD (NIN + 4)                       /* fp32 row stride: 516 */
#define ATTN_SMEM_FLOATS (32 * SF_LD + 32 * 129)
#define ATTN_SMEM_BYTES  (ATTN_SMEM_FLOATS * 4)

__global__ void __launch_bounds__(256)
attn_scores(const float* __restrict__ feats, const float* __restrict__ w_i2h,
            const float* __restrict__ hprime, const float* __restrict__ w_score,
            float* __restrict__ emit) {
    extern __shared__ float smem[];
    float (*sF)[SF_LD]  = (float (*)[SF_LD])smem;            // 32 x 516 fp32 tile
    float (*sPart)[129] = (float (*)[129])(smem + 32 * SF_LD);

    const int tid  = threadIdx.x;
    const int lane = tid & 31;
    const int wave = tid >> 5;
    const int m0   = blockIdx.x * 32;

    // ---- stage 32x512 fp32 feats tile into LDS ----
#ifdef HAVE_ASYNC_LDS
    // async DMA: 16B per issue, 4096 chunks, ASYNCcnt-tracked, no VGPR bounce
    for (int i = tid; i < 32 * (NIN / 4); i += 256) {
        const int r = i >> 7;                 // 128 vec4 chunks per row
        const int c = (i & 127) << 2;
        __builtin_amdgcn_global_load_async_to_lds_b128(
            (gbl_v4i*)(feats + (size_t)(m0 + r) * NIN + c),
            (lds_v4i*)&sF[r][c], 0, 0);
    }
    async_wait0();
#else
    for (int i = tid; i < 32 * NIN; i += 256) {
        const int r = i >> 9, c = i & (NIN - 1);
        sF[r][c] = feats[(size_t)(m0 + r) * NIN + c];
    }
#endif
    __syncthreads();

    const int n0 = wave * 64;
    v8f acc[2][4] = {};
    for (int k0 = 0; k0 < NIN; k0 += 32) {
        const float* pa0 = &sF[(lane & 15)][k0 + ((lane >> 4) << 3)];
        const float* pa1 = &sF[16 + (lane & 15)][k0 + ((lane >> 4) << 3)];
        v16bf a0 = cvt16(pa0, pa0 + 16);
        v16bf a1 = cvt16(pa1, pa1 + 16);
        v16bf b0 = load_b_frag_f32(w_i2h, NIN, n0,      k0, lane);
        v16bf b1 = load_b_frag_f32(w_i2h, NIN, n0 + 16, k0, lane);
        v16bf b2 = load_b_frag_f32(w_i2h, NIN, n0 + 32, k0, lane);
        v16bf b3 = load_b_frag_f32(w_i2h, NIN, n0 + 48, k0, lane);
        acc[0][0] = wmma_bf16(a0, b0, acc[0][0]);
        acc[1][0] = wmma_bf16(a1, b0, acc[1][0]);
        acc[0][1] = wmma_bf16(a0, b1, acc[0][1]);
        acc[1][1] = wmma_bf16(a1, b1, acc[1][1]);
        acc[0][2] = wmma_bf16(a0, b2, acc[0][2]);
        acc[1][2] = wmma_bf16(a1, b2, acc[1][2]);
        acc[0][3] = wmma_bf16(a0, b3, acc[0][3]);
        acc[1][3] = wmma_bf16(a1, b3, acc[1][3]);
    }

    // epilogue: tanh(acc + h'[b,h]) * w_score[h], partial over this wave's
    // 64 h columns, then cross-wave/lane reduce through LDS.
    const int nc = lane & 15, mh = lane >> 4;
    const int bbase = m0 & (BSZ - 1);   // tile lies within one t
#pragma unroll
    for (int mt = 0; mt < 2; ++mt) {
#pragma unroll
        for (int r = 0; r < 8; ++r) {
            const int mloc = mt * 16 + mh * 8 + r;
            const int b = bbase + mloc;
            float s = 0.0f;
#pragma unroll
            for (int nt = 0; nt < 4; ++nt) {
                const int hcol = n0 + nt * 16 + nc;
                const float v = acc[mt][nt][r] + hprime[(size_t)b * NH + hcol];
                s += tanhf(v) * w_score[hcol];
            }
            sPart[mloc][wave * 16 + nc] = s;
        }
    }
    __syncthreads();

    if (tid < 32) {
        float s = 0.0f;
        for (int j = 0; j < 128; ++j) s += sPart[tid][j];
        emit[m0 + tid] = s;
    }
}

// ---------------------------------------------------------------------------
// Softmax over T (axis 0) per column b. One block per b, 256 threads over t.
// ---------------------------------------------------------------------------
__global__ void __launch_bounds__(256)
softmax_t(const float* __restrict__ emit, float* __restrict__ alpha) {
    __shared__ float red[256];
    const int b = blockIdx.x, t = threadIdx.x;
    const float e = emit[t * BSZ + b];

    red[t] = e;
    __syncthreads();
    for (int s = 128; s > 0; s >>= 1) {
        if (t < s) red[t] = fmaxf(red[t], red[t + s]);
        __syncthreads();
    }
    const float mx = red[0];
    __syncthreads();

    const float ex = __expf(e - mx);
    red[t] = ex;
    __syncthreads();
    for (int s = 128; s > 0; s >>= 1) {
        if (t < s) red[t] += red[t + s];
        __syncthreads();
    }
    alpha[t * BSZ + b] = ex / red[0];
}

// ---------------------------------------------------------------------------
// context[b,c] = sum_t alpha[t,b] * feats[t,b,c]; also builds x=[context|emb].
// One block per b; 256 threads cover c and c+256. feats read coalesced.
// ---------------------------------------------------------------------------
__global__ void __launch_bounds__(256)
context_x(const float* __restrict__ feats, const float* __restrict__ alpha,
          const float* __restrict__ cur_embed, float* __restrict__ x) {
    const int b = blockIdx.x, c = threadIdx.x;
    float a0 = 0.0f, a1 = 0.0f;
    for (int t = 0; t < TSZ; ++t) {
        const float al = alpha[t * BSZ + b];
        const float* f = feats + ((size_t)t * BSZ + b) * NIN;
        a0 += al * f[c];
        a1 += al * f[c + 256];
    }
    float* xr = x + (size_t)b * NX;
    xr[c]        = a0;
    xr[c + 256]  = a1;
    xr[NIN + c]  = cur_embed[(size_t)b * NEMB + c];
}

// ---------------------------------------------------------------------------
// GRU elementwise tail: r,z,n gates -> nh.  i over B*NH.
// ---------------------------------------------------------------------------
__global__ void __launch_bounds__(256)
gru_out(const float* __restrict__ gi, const float* __restrict__ gh,
        const float* __restrict__ h, float* __restrict__ nh) {
    const int i = blockIdx.x * 256 + threadIdx.x;
    const int b = i >> 9, hc = i & (NH - 1);
    const float* gib = gi + (size_t)b * N3H;
    const float* ghb = gh + (size_t)b * N3H;
    const float r = 1.0f / (1.0f + __expf(-(gib[hc] + ghb[hc])));
    const float z = 1.0f / (1.0f + __expf(-(gib[NH + hc] + ghb[NH + hc])));
    const float n = tanhf(gib[2 * NH + hc] + r * ghb[2 * NH + hc]);
    nh[i] = (1.0f - z) * n + z * h[i];
}

// ---------------------------------------------------------------------------
extern "C" void kernel_launch(void* const* d_in, const int* in_sizes, int n_in,
                              void* d_out, int out_size, void* d_ws, size_t ws_size,
                              hipStream_t stream) {
    (void)in_sizes; (void)n_in; (void)out_size; (void)ws_size;
    const float* feats     = (const float*)d_in[0];
    const float* h         = (const float*)d_in[1];
    const float* cur_embed = (const float*)d_in[2];
    const float* w_i2h     = (const float*)d_in[3];
    const float* w_h2h     = (const float*)d_in[4];
    const float* b_h2h     = (const float*)d_in[5];
    const float* w_score   = (const float*)d_in[6];
    const float* w_ih      = (const float*)d_in[7];
    const float* w_hh      = (const float*)d_in[8];
    const float* b_ih      = (const float*)d_in[9];
    const float* b_hh      = (const float*)d_in[10];

    float* out       = (float*)d_out;
    float* nh_out    = out;                 // [B, NH]
    float* alpha_out = out + BSZ * NH;      // [T, B]

    float* ws     = (float*)d_ws;
    float* hprime = ws;  ws += BSZ * NH;    // [B, NH]
    float* gh     = ws;  ws += BSZ * N3H;   // [B, 3H]
    float* gi     = ws;  ws += BSZ * N3H;   // [B, 3H]
    float* emit   = ws;  ws += TSZ * BSZ;   // [T, B]
    float* x      = ws;  ws += BSZ * NX;    // [B, NIN+NEMB]

    // h' = h @ w_h2h^T + b_h2h          (256x512x512)
    gemm_at_wt<<<dim3(BSZ / 32, NH / 512), 256, 0, stream>>>(h, w_h2h, b_h2h, hprime, NH, NH);
    // gh = h @ w_hh^T + b_hh            (256x1536x512)
    gemm_at_wt<<<dim3(BSZ / 32, N3H / 512), 256, 0, stream>>>(h, w_hh, b_hh, gh, N3H, NH);
    // emit (fused big GEMM + tanh + score-dot)   (65536x512x512)
    attn_scores<<<(TSZ * BSZ) / 32, 256, ATTN_SMEM_BYTES, stream>>>(feats, w_i2h, hprime,
                                                                   w_score, emit);
    // alpha = softmax_T(emit) -> directly into d_out tail
    softmax_t<<<BSZ, 256, 0, stream>>>(emit, alpha_out);
    // context + x = [context | cur_embed]
    context_x<<<BSZ, 256, 0, stream>>>(feats, alpha_out, cur_embed, x);
    // gi = x @ w_ih^T + b_ih            (256x1536x768)
    gemm_at_wt<<<dim3(BSZ / 32, N3H / 512), 256, 0, stream>>>(x, w_ih, b_ih, gi, N3H, NX);
    // GRU tail -> nh
    gru_out<<<(BSZ * NH) / 256, 256, 0, stream>>>(gi, gh, h, nh_out);
}
